// NonLocalBlock_26147760898421
// MI455X (gfx1250) — matrix operand order repaired and checked by reading, not compiled, not run on hardware
//
#include <hip/hip_runtime.h>
#include <hip/hip_bf16.h>
#include <math.h>

typedef __attribute__((ext_vector_type(16))) __bf16 v16bf;
typedef __attribute__((ext_vector_type(8)))  __bf16 v8bf;
typedef __attribute__((ext_vector_type(8)))  float  v8f;
typedef __attribute__((ext_vector_type(4)))  float  v4f;
typedef __attribute__((ext_vector_type(4)))  unsigned int u32x4;
typedef __attribute__((ext_vector_type(8)))  int    i32x8;
typedef __attribute__((ext_vector_type(4)))  int    i32x4;

#define IN_C  256
#define INTER 128
#define NB    8
#define NPOS  4096

#ifndef __has_builtin
#define __has_builtin(x) 0
#endif
#if __has_builtin(__builtin_amdgcn_tensor_load_to_lds) && \
    __has_builtin(__builtin_amdgcn_s_wait_tensorcnt)
#define USE_TDM 1
#else
#define USE_TDM 0
#endif

// Native fptrunc (RTNE): lets the backend pick the hardware bf16 convert
// (packed v_cvt_pk_bf16_f32-class ops) instead of a 3-VALU integer expansion.
static __device__ __forceinline__ __bf16 f2bf(float f) {
    return (__bf16)f;
}

static __device__ __forceinline__ v8f wmma_bf16(v16bf a, v16bf b, v8f c) {
    return __builtin_amdgcn_wmma_f32_16x16x32_bf16(false, a, false, b, (short)0, c, false, false);
}

// A-operand (16x32 bf16): lane holds K = {c0..c0+7, c0+16..c0+23}, c0 = (lane<16 ? 0 : 8)
static __device__ __forceinline__ v16bf load_a_bf16(const __bf16* row, int c0) {
    v8bf lo = *(const v8bf*)(row + c0);
    v8bf hi = *(const v8bf*)(row + c0 + 16);
    v16bf r;
#pragma unroll
    for (int e = 0; e < 8; ++e) { r[e] = lo[e]; r[e + 8] = hi[e]; }
    return r;
}

// B-operand (32x16 bf16): lane = N-column, K = {c0..c0+7, c0+16..c0+23} strided by ld
static __device__ __forceinline__ v16bf load_b_bf16(const __bf16* p, int ld, int c0) {
    v16bf r;
#pragma unroll
    for (int e = 0; e < 8; ++e) { r[e] = p[(size_t)(c0 + e) * ld]; r[e + 8] = p[(size_t)(c0 + 16 + e) * ld]; }
    return r;
}

// A-operand built from 32-bit float memory (weights)
static __device__ __forceinline__ v16bf load_a_f32(const float* row, int c0) {
    v4f f0 = *(const v4f*)(row + c0);
    v4f f1 = *(const v4f*)(row + c0 + 4);
    v4f f2 = *(const v4f*)(row + c0 + 16);
    v4f f3 = *(const v4f*)(row + c0 + 20);
    v16bf r;
#pragma unroll
    for (int e = 0; e < 4; ++e) {
        r[e]      = f2bf(f0[e]);
        r[e + 4]  = f2bf(f1[e]);
        r[e + 8]  = f2bf(f2[e]);
        r[e + 12] = f2bf(f3[e]);
    }
    return r;
}

static __device__ __forceinline__ float redmax16(float v) {
#pragma unroll
    for (int off = 8; off >= 1; off >>= 1)
        v = fmaxf(v, __shfl_xor(v, off, 32));
    return v;
}
static __device__ __forceinline__ float redsum16(float v) {
#pragma unroll
    for (int off = 8; off >= 1; off >>= 1)
        v += __shfl_xor(v, off, 32);
    return v;
}

#if USE_TDM
// Issue one TDM 2D tile load (global -> LDS) per the CDNA5 D# layout (ISA ch.8):
//   group0: count=1, lds_addr, 57b global_addr, type=2
//   group1: data_size=2B, pad_enable, pad_interval/amount, dims/strides
static __device__ __forceinline__ void tdm_load_2d(
    unsigned lds_off, const void* gptr,
    unsigned tensor_d0, unsigned tensor_d1,
    unsigned tile_d0, unsigned tile_d1,
    unsigned long long stride0_elems,
    int pad_interval_code, int pad_amount_code)
{
    unsigned long long ga = (unsigned long long)(size_t)gptr;
    u32x4 g0;
    g0[0] = 1u;                                        // count=1, user desc
    g0[1] = lds_off;                                   // LDS byte address
    g0[2] = (unsigned)(ga & 0xffffffffu);              // global_addr[31:0]
    g0[3] = (unsigned)((ga >> 32) & 0x01ffffffu)       // global_addr[56:32]
          | (2u << 30);                                // type = 2 ("image")
    i32x8 g1;
    g1[0] = (int)((1u << 16)                           // data_size = 2 bytes
          | (1u << 20)                                 // pad_enable
          | ((unsigned)pad_interval_code << 22)
          | ((unsigned)pad_amount_code << 25));
    g1[1] = (int)((tensor_d0 & 0xffffu) << 16);        // dim0[15:0] (barrier addr=0)
    g1[2] = (int)((tensor_d0 >> 16) | ((tensor_d1 & 0xffffu) << 16));
    g1[3] = (int)(((tensor_d1 >> 16) & 0xffffu) | ((tile_d0 & 0xffffu) << 16));
    g1[4] = (int)(tile_d1 & 0xffffu);                  // tile_dim1, tile_dim2=0
    g1[5] = (int)(stride0_elems & 0xffffffffull);      // dim0_stride[31:0]
    g1[6] = (int)((stride0_elems >> 32) & 0xffffull);  // dim0_stride[47:32]
    g1[7] = 0;
    i32x4 z4 = {0, 0, 0, 0};
#if __clang_major__ >= 23
    i32x8 z8 = {0, 0, 0, 0, 0, 0, 0, 0};
    __builtin_amdgcn_tensor_load_to_lds(g0, g1, z4, z4, z8, 0);
#else
    __builtin_amdgcn_tensor_load_to_lds(g0, g1, z4, z4, 0);
#endif
}
#endif

// ---------------------------------------------------------------------------
// Kernel 1: fused g/theta/phi 1x1-conv projections (bf16 WMMA)
// g_ws, th_ws: position-major [b][q][128]; ph_ws: channel-major [b][c][q]
// ---------------------------------------------------------------------------
__global__ __launch_bounds__(256) void proj_kernel(
    const float* __restrict__ x,
    const float* __restrict__ g_w,  const float* __restrict__ g_b,
    const float* __restrict__ th_w, const float* __restrict__ th_b,
    const float* __restrict__ ph_w, const float* __restrict__ ph_b,
    __bf16* __restrict__ g_ws, __bf16* __restrict__ th_ws, __bf16* __restrict__ ph_ws)
{
    const int b    = blockIdx.y;
    const int qb   = blockIdx.x * 64;
    const int t    = threadIdx.x;
    const int wv   = t >> 5;
    const int lane = t & 31;
    const int l16  = lane & 15;
    const int hi   = lane >> 4;
    const int c0   = hi ? 8 : 0;

    __shared__ __align__(16) __bf16 xs[IN_C][72];   // x tile, padded rows

    // Stage x[b][c][qb..qb+63] -> LDS bf16 (row per thread)
    {
        const float* xrow = x + ((size_t)b * IN_C + t) * NPOS + qb;
#pragma unroll
        for (int j = 0; j < 64; j += 8) {
            v4f a0 = *(const v4f*)(xrow + j);
            v4f a1 = *(const v4f*)(xrow + j + 4);
            v8bf o;
#pragma unroll
            for (int e = 0; e < 4; ++e) { o[e] = f2bf(a0[e]); o[e + 4] = f2bf(a1[e]); }
            *(v8bf*)(&xs[t][j]) = o;
        }
    }
    __syncthreads();

    const int ib   = wv * 16;
    const int irow = ib + l16;

    // One projection; `which` is a compile-time constant at each call site.
    auto do_proj = [&](const float* __restrict__ W, const float* __restrict__ BIAS,
                       __bf16* __restrict__ dst, int which) {
        v16bf A[8];
#pragma unroll
        for (int ks = 0; ks < 8; ++ks)
            A[ks] = load_a_f32(W + irow * IN_C + ks * 32, c0);
        float bias[8];
#pragma unroll
        for (int r = 0; r < 8; ++r)
            bias[r] = BIAS[ib + r + 8 * hi];

#pragma unroll
        for (int nt = 0; nt < 4; ++nt) {
            v8f acc = {0.f, 0.f, 0.f, 0.f, 0.f, 0.f, 0.f, 0.f};
#pragma unroll
            for (int ks = 0; ks < 8; ++ks) {
                v16bf Bv = load_b_bf16(&xs[ks * 32][nt * 16 + l16], 72, c0);
                acc = wmma_bf16(A[ks], Bv, acc);
            }
            const int q = qb + nt * 16 + l16;
#pragma unroll
            for (int r = 0; r < 8; ++r) {
                const int i = ib + r + 8 * hi;
                __bf16 bv = f2bf(acc[r] + bias[r]);
                if (which == 2)   // phi: channel-major
                    dst[(size_t)b * INTER * NPOS + (size_t)i * NPOS + q] = bv;
                else              // g/theta: position-major
                    dst[(size_t)b * NPOS * INTER + (size_t)q * INTER + i] = bv;
            }
        }
    };

    do_proj(g_w,  g_b,  g_ws,  0);
    do_proj(th_w, th_b, th_ws, 1);
    do_proj(ph_w, ph_b, ph_ws, 2);
}

// ---------------------------------------------------------------------------
// Kernel 2: flash-attention style  y = softmax(theta^T phi) g^T  (bf16 WMMA)
// 8 waves x 16 queries = 128 q per block; 32-key tiles double-buffered in LDS
// via the Tensor Data Mover (wave 0 issues TDM, s_wait_tensorcnt + barrier).
// y_ws written channel-major [b][i][q]
// ---------------------------------------------------------------------------
__global__ __launch_bounds__(256) void attn_kernel(
    const __bf16* __restrict__ th_ws, const __bf16* __restrict__ ph_ws,
    const __bf16* __restrict__ g_ws,  __bf16* __restrict__ y_ws)
{
    const int b    = blockIdx.y;
    const int qb   = blockIdx.x * 128;
    const int t    = threadIdx.x;
    const int wv   = t >> 5;
    const int lane = t & 31;
    const int l16  = lane & 15;
    const int hi   = lane >> 4;
    const int c0   = hi ? 8 : 0;

    // Double-buffered key tiles. Rows padded: 64B data + 16B pad (phi),
    // 256B data + 16B pad (g) -> matches TDM pad_interval/pad_amount codes.
    __shared__ __align__(16) __bf16 phs[2][INTER][40];   // [c][32 keys]
    __shared__ __align__(16) __bf16 gs[2][32][136];      // [k][128 i]
    __shared__ __align__(16) __bf16 ps[8][16][40];       // per-wave P tile [q][32 k]

    const int qw = qb + wv * 16;

    // Preload A_theta: rows q = qw + l16, K = 128 channels (4 K-steps)
    v16bf Ath[4];
    {
        const __bf16* trow = th_ws + ((size_t)b * NPOS + qw + l16) * INTER;
#pragma unroll
        for (int ks = 0; ks < 4; ++ks)
            Ath[ks] = load_a_bf16(trow + ks * 32, c0);
    }

    v8f y[8];
    const v8f vz = {0.f, 0.f, 0.f, 0.f, 0.f, 0.f, 0.f, 0.f};
#pragma unroll
    for (int it = 0; it < 8; ++it) y[it] = vz;
    float m[8], lsum[8];
#pragma unroll
    for (int r = 0; r < 8; ++r) { m[r] = -INFINITY; lsum[r] = 0.f; }

    // ---- tile staging: TDM (async DMA) or manual fallback, double-buffered
    auto stage = [&](int buf, int kb) {
#if USE_TDM
        if (wv == 0) {
            // phi tile: 128 rows x 32 keys, row stride 4096 elems
            tdm_load_2d((unsigned)(size_t)&phs[buf][0][0],
                        ph_ws + (size_t)b * INTER * NPOS + kb,
                        NPOS, INTER, 32, INTER, NPOS,
                        /*pad_interval 16 dw=64B*/ 3, /*pad 4 dw=16B*/ 3);
            // g tile: 32 rows x 128 i, row stride 128 elems
            tdm_load_2d((unsigned)(size_t)&gs[buf][0][0],
                        g_ws + ((size_t)b * NPOS + kb) * INTER,
                        INTER, NPOS, INTER, 32, INTER,
                        /*pad_interval 64 dw=256B*/ 5, /*pad 4 dw=16B*/ 3);
        }
#else
        if (t < 128) {
            const __bf16* src = ph_ws + ((size_t)b * INTER + t) * NPOS + kb;
#pragma unroll
            for (int j = 0; j < 32; j += 8)
                *(v8bf*)(&phs[buf][t][j]) = *(const v8bf*)(src + j);
        } else {
            const int tt   = t - 128;
            const int k    = tt >> 2;
            const int part = (tt & 3) * 32;
            const __bf16* src = g_ws + ((size_t)b * NPOS + kb + k) * INTER + part;
#pragma unroll
            for (int j = 0; j < 32; j += 8)
                *(v8bf*)(&gs[buf][k][part + j]) = *(const v8bf*)(src + j);
        }
#endif
    };

    stage(0, 0);
    int cur = 0;

    for (int kb = 0; kb < NPOS; kb += 32) {
#if USE_TDM
        if (wv == 0) __builtin_amdgcn_s_wait_tensorcnt(0);
#endif
        __syncthreads();                 // current tile visible to all waves
        if (kb + 32 < NPOS)
            stage(cur ^ 1, kb + 32);     // overlap next-tile DMA with compute

        // ---- S = theta . phi^T : 16 q x 32 keys, K = 128 channels
        v8f s0 = vz, s1 = vz;
#pragma unroll
        for (int ks = 0; ks < 4; ++ks) {
            const __bf16* prow = &phs[cur][ks * 32][l16];
            v16bf B0 = load_b_bf16(prow, 40, c0);
            v16bf B1 = load_b_bf16(prow + 16, 40, c0);
            s0 = wmma_bf16(Ath[ks], B0, s0);
            s1 = wmma_bf16(Ath[ks], B1, s1);
        }

        // ---- online softmax (rows live one-per-VGPR, striped over 16 lanes)
        float al[8];
#pragma unroll
        for (int r = 0; r < 8; ++r) {
            float mx = redmax16(fmaxf(s0[r], s1[r]));
            float mn = fmaxf(m[r], mx);
            float a  = __expf(m[r] - mn);
            float p0 = __expf(s0[r] - mn);
            float p1 = __expf(s1[r] - mn);
            float rs = redsum16(p0 + p1);
            lsum[r] = lsum[r] * a + rs;
            m[r]    = mn;
            al[r]   = a;
            s0[r]   = p0;
            s1[r]   = p1;
        }
#pragma unroll
        for (int it = 0; it < 8; ++it)
#pragma unroll
            for (int r = 0; r < 8; ++r) y[it][r] *= al[r];

        // ---- repack P (C-layout) -> A-layout via per-wave LDS buffer
#pragma unroll
        for (int r = 0; r < 8; ++r) {
            const int q = r + 8 * hi;
            ps[wv][q][l16]      = f2bf(s0[r]);
            ps[wv][q][16 + l16] = f2bf(s1[r]);
        }
        asm volatile("s_wait_dscnt 0" ::: "memory");
        v16bf Ap = load_a_bf16(&ps[wv][l16][0], c0);

        // ---- y += P . g : 16 q x 128 i, K = 32 keys
#pragma unroll
        for (int it = 0; it < 8; ++it) {
            v16bf Bg = load_b_bf16(&gs[cur][0][it * 16 + l16], 136, c0);
            y[it] = wmma_bf16(Ap, Bg, y[it]);
        }
        cur ^= 1;
    }

    // ---- finalize: y /= rowsum, store channel-major bf16
#pragma unroll
    for (int r = 0; r < 8; ++r) lsum[r] = 1.f / lsum[r];
#pragma unroll
    for (int it = 0; it < 8; ++it) {
        const int i = it * 16 + l16;
#pragma unroll
        for (int r = 0; r < 8; ++r) {
            const int q = qw + r + 8 * hi;
            y_ws[((size_t)b * INTER + i) * NPOS + q] = f2bf(y[it][r] * lsum[r]);
        }
    }
}

// ---------------------------------------------------------------------------
// Kernel 3: W_y = W_w . y + W_b + x   (bf16 WMMA, fused bias + residual)
// ---------------------------------------------------------------------------
__global__ __launch_bounds__(256) void out_kernel(
    const float* __restrict__ x, const float* __restrict__ W_w,
    const float* __restrict__ W_b, const __bf16* __restrict__ y_ws,
    float* __restrict__ out)
{
    const int b    = blockIdx.y;
    const int qb   = blockIdx.x * 64;
    const int t    = threadIdx.x;
    const int wv   = t >> 5;
    const int lane = t & 31;
    const int l16  = lane & 15;
    const int hi   = lane >> 4;
    const int c0   = hi ? 8 : 0;

#pragma unroll
    for (int half = 0; half < 2; ++half) {
        const int ob   = (wv + half * 8) * 16;     // 16 output channels per tile
        const int orow = ob + l16;

        v16bf A[4];
#pragma unroll
        for (int ks = 0; ks < 4; ++ks)
            A[ks] = load_a_f32(W_w + orow * INTER + ks * 32, c0);
        float bias[8];
#pragma unroll
        for (int r = 0; r < 8; ++r)
            bias[r] = W_b[ob + r + 8 * hi];

#pragma unroll
        for (int nt = 0; nt < 4; ++nt) {
            v8f acc = {0.f, 0.f, 0.f, 0.f, 0.f, 0.f, 0.f, 0.f};
#pragma unroll
            for (int ks = 0; ks < 4; ++ks) {
                const __bf16* yb =
                    y_ws + ((size_t)b * INTER + ks * 32) * NPOS + qb + nt * 16 + l16;
                v16bf Bv = load_b_bf16(yb, NPOS, c0);
                acc = wmma_bf16(A[ks], Bv, acc);
            }
            const int q = qb + nt * 16 + l16;
#pragma unroll
            for (int r = 0; r < 8; ++r) {
                const int o = ob + r + 8 * hi;
                const size_t idx = ((size_t)b * IN_C + o) * NPOS + q;
                out[idx] = acc[r] + bias[r] + x[idx];
            }
        }
    }
}

// ---------------------------------------------------------------------------
extern "C" void kernel_launch(void* const* d_in, const int* in_sizes, int n_in,
                              void* d_out, int out_size, void* d_ws, size_t ws_size,
                              hipStream_t stream) {
    (void)in_sizes; (void)n_in; (void)out_size; (void)ws_size;

    const float* x    = (const float*)d_in[0];
    const float* g_w  = (const float*)d_in[1];
    const float* g_b  = (const float*)d_in[2];
    const float* th_w = (const float*)d_in[3];
    const float* th_b = (const float*)d_in[4];
    const float* ph_w = (const float*)d_in[5];
    const float* ph_b = (const float*)d_in[6];
    const float* W_w  = (const float*)d_in[7];
    const float* W_b  = (const float*)d_in[8];
    float* out = (float*)d_out;

    char* ws = (char*)d_ws;
    const size_t seg = (size_t)NB * NPOS * INTER * sizeof(__bf16);   // 8 MB each
    __bf16* g_ws  = (__bf16*)(ws);
    __bf16* th_ws = (__bf16*)(ws + seg);
    __bf16* ph_ws = (__bf16*)(ws + 2 * seg);
    __bf16* y_ws  = (__bf16*)(ws + 3 * seg);

    proj_kernel<<<dim3(NPOS / 64, NB), 256, 0, stream>>>(
        x, g_w, g_b, th_w, th_b, ph_w, ph_b, g_ws, th_ws, ph_ws);
    attn_kernel<<<dim3(NPOS / 128, NB), 256, 0, stream>>>(th_ws, ph_ws, g_ws, y_ws);
    out_kernel<<<dim3(NPOS / 64, NB), 256, 0, stream>>>(x, W_w, W_b, y_ws, out);
}
